// SensorGCN_4131758539434
// MI455X (gfx1250) — compile-verified, matrix operand order checked by reading, not run on hardware
//
#include <hip/hip_runtime.h>
#include <hip/hip_bf16.h>

#define N_NODES 100000
#define N_EDGES 1000000
#define F_INC   32
#define HID     64
#define N_CLS   3
#define N_GRAPH 64
#define BN_EPS  1e-5f

typedef __attribute__((ext_vector_type(16))) _Float16 v16h;
typedef __attribute__((ext_vector_type(8)))  float    v8f;

// ---------------- utility kernels ----------------

__global__ void set_kernel(float* __restrict__ p, float v, int n) {
    int i = blockIdx.x * blockDim.x + threadIdx.x;
    if (i < n) p[i] = v;
}

// deg accumulation: dinv[] pre-set to 1.0 (self loop), add 1 per incoming edge
__global__ void deg_kernel(const int* __restrict__ dst, float* __restrict__ deg, int e) {
    int i = blockIdx.x * blockDim.x + threadIdx.x;
    if (i < e) atomicAdd(&deg[dst[i]], 1.0f);
}

__global__ void rsqrt_kernel(float* __restrict__ d, int n) {
    int i = blockIdx.x * blockDim.x + threadIdx.x;
    if (i < n) d[i] = rsqrtf(d[i]);   // deg >= 1 always (self loop)
}

// ---------------- WMMA GEMM: out[N x 64] = in[N x K] @ W[K x 64] ----------------
// one wave computes a 16-row x 64-col strip; 4 accumulators; A reused across col tiles.
__global__ void gemm_wmma(const float* __restrict__ in, const float* __restrict__ W,
                          float* __restrict__ out, int K) {
    const int lane   = threadIdx.x & 31;
    const int wave   = threadIdx.x >> 5;
    const int rowTile = blockIdx.x * 4 + wave;
    const int rowBase = rowTile * 16;
    if (rowBase >= N_NODES) return;          // uniform per wave

    const int m   = lane & 15;               // row (A) / col (B/C/D) within tile
    const int khi = (lane >> 4) << 3;        // 0 or 8: lane-group K offset
    v8f acc0 = {}, acc1 = {}, acc2 = {}, acc3 = {};

    for (int kb = 0; kb < K; kb += 32) {
        // A tile 16x32 f16: half i -> K = kb + khi + (i<8 ? i : 16+i-8)
        v16h a;
        const float* arow = in + (size_t)(rowBase + m) * K + kb + khi;
        #pragma unroll
        for (int i = 0; i < 8; ++i) {
            a[i]     = (_Float16)arow[i];
            a[i + 8] = (_Float16)arow[i + 16];
        }
        #pragma unroll
        for (int ct = 0; ct < 4; ++ct) {
            // B tile 32x16 f16: same K mapping, column = ct*16 + m
            v16h b;
            const float* wcol = W + (size_t)(kb + khi) * HID + ct * 16 + m;
            #pragma unroll
            for (int i = 0; i < 8; ++i) {
                b[i]     = (_Float16)wcol[(size_t)i * HID];
                b[i + 8] = (_Float16)wcol[(size_t)(i + 16) * HID];
            }
            v8f c = (ct == 0) ? acc0 : (ct == 1) ? acc1 : (ct == 2) ? acc2 : acc3;
            c = __builtin_amdgcn_wmma_f32_16x16x32_f16(false, a, false, b,
                                                       (short)0, c, false, false);
            if (ct == 0) acc0 = c; else if (ct == 1) acc1 = c;
            else if (ct == 2) acc2 = c; else acc3 = c;
        }
    }
    // C/D layout: VGPR v -> row = rowBase + v + (lane>=16 ? 8 : 0), col = ct*16 + m
    const int rowOff = (lane >> 4) << 3;
    #pragma unroll
    for (int v = 0; v < 8; ++v) {
        size_t r = (size_t)(rowBase + rowOff + v) * HID;
        out[r +  0 + m] = acc0[v];
        out[r + 16 + m] = acc1[v];
        out[r + 32 + m] = acc2[v];
        out[r + 48 + m] = acc3[v];
    }
}

// ---------------- graph aggregation ----------------

// self-loop term initializes agg: agg[i,f] = dinv[i]^2 * h[i,f]
__global__ void selfloop_kernel(const float* __restrict__ h, const float* __restrict__ dinv,
                                float* __restrict__ agg) {
    size_t idx = (size_t)blockIdx.x * blockDim.x + threadIdx.x;
    if (idx >= (size_t)N_NODES * HID) return;
    float di = dinv[idx >> 6];
    agg[idx] = di * di * h[idx];
}

// per-(edge,feature) scatter-add; atomics resolve in 192MB L2 (working set fits)
__global__ void edge_scatter(const float* __restrict__ h, const int* __restrict__ src,
                             const int* __restrict__ dst, const float* __restrict__ dinv,
                             float* __restrict__ agg) {
    size_t idx = (size_t)blockIdx.x * blockDim.x + threadIdx.x;
    if (idx >= (size_t)N_EDGES * HID) return;
    int e = (int)(idx >> 6);
    int f = (int)(idx & 63);
    int s = src[e], d = dst[e];
    float nrm = dinv[s] * dinv[d];
    atomicAdd(&agg[(size_t)d * HID + f], nrm * h[(size_t)s * HID + f]);
}

// ---------------- BatchNorm ----------------

// hierarchical per-feature sum / sum-of-squares (bias folded into the statistic)
__global__ void bn_stats(const float* __restrict__ h, const float* __restrict__ bias,
                         float* __restrict__ sums, float* __restrict__ sumsq) {
    __shared__ float ssum[HID], ssq[HID];
    const int f = threadIdx.x & 63;
    float s = 0.f, q = 0.f;
    const int rowsPerBlk = blockDim.x >> 6;
    float bf = bias[f];
    for (size_t i = (size_t)blockIdx.x * rowsPerBlk + (threadIdx.x >> 6);
         i < N_NODES; i += (size_t)gridDim.x * rowsPerBlk) {
        float v = h[i * HID + f] + bf;
        s += v; q += v * v;
    }
    if (threadIdx.x < HID) { ssum[threadIdx.x] = 0.f; ssq[threadIdx.x] = 0.f; }
    __syncthreads();
    atomicAdd(&ssum[f], s);
    atomicAdd(&ssq[f], q);
    __syncthreads();
    if (threadIdx.x < HID) {
        atomicAdd(&sums[threadIdx.x], ssum[threadIdx.x]);
        atomicAdd(&sumsq[threadIdx.x], ssq[threadIdx.x]);
    }
}

__global__ void bn_apply(float* __restrict__ h, const float* __restrict__ bias,
                         const float* __restrict__ sums, const float* __restrict__ sumsq,
                         const float* __restrict__ gamma, const float* __restrict__ beta,
                         int relu) {
    size_t idx = (size_t)blockIdx.x * blockDim.x + threadIdx.x;
    if (idx >= (size_t)N_NODES * HID) return;
    int f = (int)(idx & 63);
    float mean = sums[f] * (1.0f / N_NODES);
    float var  = sumsq[f] * (1.0f / N_NODES) - mean * mean;
    float v = (h[idx] + bias[f] - mean) * rsqrtf(var + BN_EPS) * gamma[f] + beta[f];
    if (relu) v = fmaxf(v, 0.f);
    h[idx] = v;
}

// ---------------- pooling + MLP head ----------------

__global__ void pool_kernel(const float* __restrict__ h, const int* __restrict__ batch,
                            float* __restrict__ pool, float* __restrict__ cnt) {
    size_t idx = (size_t)blockIdx.x * blockDim.x + threadIdx.x;
    if (idx >= (size_t)N_NODES * HID) return;
    int node = (int)(idx >> 6);
    int f    = (int)(idx & 63);
    int g    = batch[node];
    atomicAdd(&pool[(size_t)g * HID + f], h[idx]);
    if (f == 0) atomicAdd(&cnt[g], 1.0f);
}

__global__ void mlp1_kernel(const float* __restrict__ pool, const float* __restrict__ cnt,
                            const float* __restrict__ Wl1, const float* __restrict__ bl1,
                            float* __restrict__ z) {
    int idx = blockIdx.x * blockDim.x + threadIdx.x;
    if (idx >= N_GRAPH * (HID / 2)) return;
    int g = idx >> 5;
    int j = idx & 31;
    float inv = 1.0f / fmaxf(cnt[g], 1.0f);
    float acc = bl1[j];
    #pragma unroll
    for (int k = 0; k < HID; ++k)
        acc += pool[g * HID + k] * inv * Wl1[k * (HID / 2) + j];
    z[idx] = fmaxf(acc, 0.f);
}

__global__ void mlp2_kernel(const float* __restrict__ z, const float* __restrict__ Wl2,
                            const float* __restrict__ bl2, float* __restrict__ out) {
    int g = blockIdx.x * blockDim.x + threadIdx.x;
    if (g >= N_GRAPH) return;
    float logit[N_CLS];
    #pragma unroll
    for (int c = 0; c < N_CLS; ++c) {
        float acc = bl2[c];
        for (int j = 0; j < HID / 2; ++j)
            acc += z[g * (HID / 2) + j] * Wl2[j * N_CLS + c];
        logit[c] = acc;
    }
    float mx = fmaxf(logit[0], fmaxf(logit[1], logit[2]));
    float e0 = __expf(logit[0] - mx), e1 = __expf(logit[1] - mx), e2 = __expf(logit[2] - mx);
    float inv = 1.0f / (e0 + e1 + e2);
    out[g * N_CLS + 0] = e0 * inv;
    out[g * N_CLS + 1] = e1 * inv;
    out[g * N_CLS + 2] = e2 * inv;
}

// ---------------- launcher ----------------

extern "C" void kernel_launch(void* const* d_in, const int* in_sizes, int n_in,
                              void* d_out, int out_size, void* d_ws, size_t ws_size,
                              hipStream_t stream) {
    (void)in_sizes; (void)n_in; (void)out_size; (void)ws_size;

    const float* x     = (const float*)d_in[0];
    const int*   ei    = (const int*)d_in[1];
    const int*   batch = (const int*)d_in[2];
    const float* W1  = (const float*)d_in[3];  const float* b1  = (const float*)d_in[4];
    const float* g1  = (const float*)d_in[5];  const float* be1 = (const float*)d_in[6];
    const float* W2  = (const float*)d_in[7];  const float* b2  = (const float*)d_in[8];
    const float* g2  = (const float*)d_in[9];  const float* be2 = (const float*)d_in[10];
    const float* W3  = (const float*)d_in[11]; const float* b3  = (const float*)d_in[12];
    const float* g3  = (const float*)d_in[13]; const float* be3 = (const float*)d_in[14];
    const float* Wl1 = (const float*)d_in[15]; const float* bl1 = (const float*)d_in[16];
    const float* Wl2 = (const float*)d_in[17]; const float* bl2 = (const float*)d_in[18];

    const int* src = ei;
    const int* dst = ei + N_EDGES;

    float* ws   = (float*)d_ws;
    float* dinv = ws;                                    // N
    float* bufA = dinv + N_NODES;                        // N*64 (gemm out)
    float* bufB = bufA + (size_t)N_NODES * HID;          // N*64 (aggregated / layer out)
    float* stat = bufB + (size_t)N_NODES * HID;          // 128 (sum | sumsq)
    float* pool = stat + 2 * HID;                        // G*64
    float* cnt  = pool + N_GRAPH * HID;                  // G
    float* z    = cnt + N_GRAPH;                         // G*32

    const size_t NH = (size_t)N_NODES * HID;             // 6.4M
    const size_t EH = (size_t)N_EDGES * HID;             // 64M
    const int T = 256;
    dim3 blk(T);

    // --- symmetric normalization (self-loops folded via deg init = 1) ---
    set_kernel  <<<(N_NODES + T - 1) / T, blk, 0, stream>>>(dinv, 1.0f, N_NODES);
    deg_kernel  <<<(N_EDGES + T - 1) / T, blk, 0, stream>>>(dst, dinv, N_EDGES);
    rsqrt_kernel<<<(N_NODES + T - 1) / T, blk, 0, stream>>>(dinv, N_NODES);

    const int rowTiles   = N_NODES / 16;                 // 6250
    const int gemmBlocks = (rowTiles + 3) / 4;

    auto layer = [&](const float* in, int K, const float* W, const float* b,
                     const float* g, const float* be, int relu) {
        gemm_wmma      <<<gemmBlocks, dim3(128), 0, stream>>>(in, W, bufA, K);
        selfloop_kernel<<<(unsigned)((NH + T - 1) / T), blk, 0, stream>>>(bufA, dinv, bufB);
        edge_scatter   <<<(unsigned)((EH + T - 1) / T), blk, 0, stream>>>(bufA, src, dst, dinv, bufB);
        set_kernel     <<<1, 128, 0, stream>>>(stat, 0.f, 2 * HID);
        bn_stats       <<<1024, blk, 0, stream>>>(bufB, b, stat, stat + HID);
        bn_apply       <<<(unsigned)((NH + T - 1) / T), blk, 0, stream>>>(bufB, b, stat, stat + HID, g, be, relu);
    };

    layer(x,    F_INC, W1, b1, g1, be1, 1);
    layer(bufB, HID,   W2, b2, g2, be2, 1);
    layer(bufB, HID,   W3, b3, g3, be3, 0);

    // --- global mean pool + MLP head + softmax ---
    set_kernel <<<(N_GRAPH * HID + N_GRAPH + T - 1) / T, blk, 0, stream>>>(pool, 0.f, N_GRAPH * HID + N_GRAPH);
    pool_kernel<<<(unsigned)((NH + T - 1) / T), blk, 0, stream>>>(bufB, batch, pool, cnt);
    mlp1_kernel<<<(N_GRAPH * (HID / 2) + T - 1) / T, blk, 0, stream>>>(pool, cnt, Wl1, bl1, z);
    mlp2_kernel<<<1, N_GRAPH, 0, stream>>>(z, Wl2, bl2, (float*)d_out);
}
